// MultiOmixGCN_18159121728097
// MI455X (gfx1250) — compile-verified, matrix operand order checked by reading, not run on hardware
//
#include <hip/hip_runtime.h>
#include <climits>

#define NN   50000
#define EE   600000
#define EPN  150000
#define HH   128
#define GG   8
#define NLAY 3
#define MT   64        // GEMM M-tile rows per block
#define PAD  136       // LDS row stride in halves (conflict-free ds_load_b128)

typedef __attribute__((ext_vector_type(16))) _Float16 v16h;
typedef __attribute__((ext_vector_type(8)))  float    v8f;

// ---- ordered-int encoding for float atomicMax ----
__device__ __forceinline__ int f2o(float f) {
    int i = __float_as_int(f);
    return i >= 0 ? i : (i ^ 0x7fffffff);
}
__device__ __forceinline__ float o2f(int o) {
    int i = o >= 0 ? o : (o ^ 0x7fffffff);
    return __int_as_float(i);
}

// ---------------- utility kernels ----------------
__global__ void zero_f_kernel(float* p, int n) {
    int gid = blockIdx.x * 256 + threadIdx.x;
    if (gid < n) p[gid] = 0.f;
}

__global__ void set_mask_kernel(const int* __restrict__ pdst, float* __restrict__ mask) {
    int e = blockIdx.x * 256 + threadIdx.x;
    if (e < EPN) mask[pdst[e]] = 1.0f;
}

__global__ void enc_kernel(const float* __restrict__ x, const float* __restrict__ W,
                           const float* __restrict__ b, float* __restrict__ h) {
    int gid = blockIdx.x * 256 + threadIdx.x;   // N*H = 6.4M < 2^31
    if (gid >= NN * HH) return;
    int i = gid / HH, c = gid % HH;
    float acc = b[c];
#pragma unroll
    for (int k = 0; k < 3; k++) acc += x[i * 3 + k] * W[k * HH + c];
    h[gid] = acc;
}

__global__ void init_aggr_kernel(int* __restrict__ mI, float* __restrict__ num,
                                 float* __restrict__ den) {
    int gid = blockIdx.x * 256 + threadIdx.x;
    if (gid >= NN * HH) return;
    mI[gid] = INT_MIN;
    num[gid] = 0.f;
    den[gid] = 0.f;
}

// ---------------- edge passes (gene edges; edge encoder recomputed from LDS weights) ----------------
__global__ __launch_bounds__(256)
void gene_edge_kernel(int phase, const float* __restrict__ hin,
                      const float* __restrict__ eattr,
                      const float* __restrict__ We, const float* __restrict__ be,
                      const int* __restrict__ src, const int* __restrict__ dst,
                      const float* __restrict__ tptr,
                      int* __restrict__ mI, float* __restrict__ num, float* __restrict__ den) {
    __shared__ float sW[7 * HH + HH];
    int lt = threadIdx.y * HH + threadIdx.x;
    for (int i = lt; i < 7 * HH + HH; i += 256)
        sW[i] = (i < 7 * HH) ? We[i] : be[i - 7 * HH];
    __syncthreads();

    int e = blockIdx.x * 2 + threadIdx.y;   // E even; grid = E/2
    int c = threadIdx.x;
    float t = tptr[0];
    int s = src[e], d = dst[e];

    float ee = sW[7 * HH + c];
#pragma unroll
    for (int k = 0; k < 7; k++) ee += eattr[e * 7 + k] * sW[k * HH + c];

    float msg = fmaxf(hin[s * HH + c] + ee, 0.f) + 1e-7f;
    float logit = t * msg;
    int di = d * HH + c;
    if (phase == 0) {
        atomicMax(&mI[di], f2o(logit));
    } else {
        float ex = __expf(logit - o2f(mI[di]));
        atomicAdd(&den[di], ex);
        atomicAdd(&num[di], ex * msg);
    }
}

// ---------------- edge passes (pathway edges; e_emb given directly) ----------------
__global__ __launch_bounds__(256)
void pw_edge_kernel(int phase, const float* __restrict__ hin,
                    const float* __restrict__ pattr,
                    const int* __restrict__ src, const int* __restrict__ dst,
                    const float* __restrict__ tptr,
                    int* __restrict__ mI, float* __restrict__ num, float* __restrict__ den) {
    int e = blockIdx.x * 2 + threadIdx.y;   // EP even; grid = EP/2
    int c = threadIdx.x;
    float t = tptr[0];
    int s = src[e], d = dst[e];

    float msg = fmaxf(hin[s * HH + c] + pattr[e * HH + c], 0.f) + 1e-7f;
    float logit = t * msg;
    int di = d * HH + c;
    if (phase == 0) {
        atomicMax(&mI[di], f2o(logit));
    } else {
        float ex = __expf(logit - o2f(mI[di]));
        atomicAdd(&den[di], ex);
        atomicAdd(&num[di], ex * msg);
    }
}

// ---------------- WMMA node-update GEMM ----------------
// out = maybe_mask((hin + num/(den+1e-16)) @ W + b) [+ resid]
// Block = 256 threads = 8 waves. Block owns MT=64 rows; wave w owns 16 output cols.
// W staged transposed in LDS as f16 -> both A and B fragments load via ds_load_b128.
// Per wave: 4 k-steps x 4 m-subtiles = 16 v_wmma_f32_16x16x32_f16, B frag reused across m.
__global__ __launch_bounds__(256)
void gemm_update_kernel(const float* __restrict__ hin, const float* __restrict__ num,
                        const float* __restrict__ den,
                        const float* __restrict__ W, const float* __restrict__ bias,
                        const float* __restrict__ mask, const float* __restrict__ resid,
                        float* __restrict__ out) {
    __shared__ _Float16 sA[MT * PAD];     // 17.4 KB
    __shared__ _Float16 sW[HH * PAD];     // 34.8 KB, transposed: sW[n*PAD + k]
    int row0 = blockIdx.x * MT;
    int lt = threadIdx.x;

    // stage W^T (coalesced global reads; one-time strided LDS writes)
    for (int idx = lt; idx < HH * HH; idx += 256) {
        int k = idx / HH, n = idx % HH;
        sW[n * PAD + k] = (_Float16)W[idx];
    }
    // stage A = hin + aggr; clamp rows so EXEC stays all-1s through the WMMA section
    for (int idx = lt; idx < MT * HH; idx += 256) {
        int i = idx / HH, k = idx % HH;
        int row = row0 + i;
        if (row >= NN) row = NN - 1;
        int gi = row * HH + k;
        float a = hin[gi] + num[gi] / (den[gi] + 1e-16f);
        sA[i * PAD + k] = (_Float16)a;
    }
    __syncthreads();

    int wave = lt >> 5;
    int lane = lt & 31;
    int n0 = wave * 16;
    int lm = lane & 15;       // A: row M; B/C: col N
    int khalf = lane >> 4;    // K-group selector (16-bit fragment layout)

    v8f acc[4] = {};
#pragma unroll
    for (int k0 = 0; k0 < HH; k0 += 32) {
        v16h b;
#pragma unroll
        for (int hidx = 0; hidx < 16; hidx++) {
            int v = hidx >> 1;
            int grp = v >> 2;
            int vv = v & 3;
            int kk = k0 + grp * 16 + khalf * 8 + vv * 2 + (hidx & 1);
            b[hidx] = sW[(n0 + lm) * PAD + kk];
        }
#pragma unroll
        for (int m = 0; m < 4; m++) {
            v16h a;
#pragma unroll
            for (int hidx = 0; hidx < 16; hidx++) {
                int v = hidx >> 1;
                int grp = v >> 2;
                int vv = v & 3;
                int kk = k0 + grp * 16 + khalf * 8 + vv * 2 + (hidx & 1);
                a[hidx] = sA[(m * 16 + lm) * PAD + kk];
            }
            acc[m] = __builtin_amdgcn_wmma_f32_16x16x32_f16(
                /*neg_a=*/false, a, /*neg_b=*/false, b,
                /*c_mod=*/(short)0, acc[m], /*reuse_a=*/false, /*reuse_b=*/false);
        }
    }

    int col = n0 + lm;
    float bv = bias[col];
#pragma unroll
    for (int m = 0; m < 4; m++) {
#pragma unroll
        for (int v = 0; v < 8; v++) {
            int row = row0 + m * 16 + v + khalf * 8;   // C layout: VGPR v -> M=v / v+8
            if (row < NN) {
                float val = acc[m][v] + bv;
                int oi = row * HH + col;
                if (mask) val = mask[row] * val + resid[oi];
                out[oi] = val;
            }
        }
    }
}

// ---------------- layernorm (+optional relu); one node per 128-thread block ----------------
__global__ __launch_bounds__(128)
void ln_kernel(const float* __restrict__ hin, float* __restrict__ hout,
               const float* __restrict__ gamma, const float* __restrict__ beta, int do_relu) {
    __shared__ float rs[HH], rq[HH];
    int i = blockIdx.x, c = threadIdx.x;
    float v = hin[i * HH + c];
    rs[c] = v;
    rq[c] = v * v;
    __syncthreads();
    for (int s = HH / 2; s > 0; s >>= 1) {
        if (c < s) { rs[c] += rs[c + s]; rq[c] += rq[c + s]; }
        __syncthreads();
    }
    float mu = rs[0] * (1.f / HH);
    float var = rq[0] * (1.f / HH) - mu * mu;
    float r = rsqrtf(var + 1e-5f);
    float o = (v - mu) * r * gamma[c] + beta[c];
    if (do_relu) o = fmaxf(o, 0.f);
    hout[i * HH + c] = o;
}

// ---------------- global mean pool ----------------
__global__ __launch_bounds__(128)
void pool_kernel(const float* __restrict__ hfin, const int* __restrict__ batch,
                 float* __restrict__ outp, float* __restrict__ cnt) {
    int i = blockIdx.x, c = threadIdx.x;
    int g = batch[i];
    atomicAdd(&outp[g * HH + c], hfin[i * HH + c]);
    if (c == 0) atomicAdd(&cnt[g], 1.f);
}

__global__ void pool_div_kernel(float* __restrict__ outp, const float* __restrict__ cnt) {
    int gid = blockIdx.x * 256 + threadIdx.x;
    if (gid >= GG * HH) return;
    outp[gid] /= fmaxf(cnt[gid / HH], 1.f);
}

// ---------------- host orchestration ----------------
extern "C" void kernel_launch(void* const* d_in, const int* in_sizes, int n_in,
                              void* d_out, int out_size, void* d_ws, size_t ws_size,
                              hipStream_t stream) {
    (void)in_sizes; (void)n_in; (void)out_size; (void)ws_size;
    const float* x          = (const float*)d_in[0];
    const float* edge_attr  = (const float*)d_in[1];
    const float* p_edge_attr= (const float*)d_in[2];
    const float* W_enc      = (const float*)d_in[3];
    const float* b_enc      = (const float*)d_in[4];
    const float* W_edge     = (const float*)d_in[5];
    const float* b_edge     = (const float*)d_in[6];
    const float* Wg         = (const float*)d_in[7];
    const float* bg         = (const float*)d_in[8];
    const float* tg         = (const float*)d_in[9];
    const float* Wp         = (const float*)d_in[10];
    const float* bp         = (const float*)d_in[11];
    const float* tp         = (const float*)d_in[12];
    const float* gamma      = (const float*)d_in[13];
    const float* beta       = (const float*)d_in[14];
    const int*   eidx       = (const int*)d_in[15];
    const int*   pidx       = (const int*)d_in[16];
    const int*   batch      = (const int*)d_in[17];
    const int* src  = eidx,  *dst  = eidx + EE;
    const int* psrc = pidx,  *pdst = pidx + EPN;

    size_t NH = (size_t)NN * HH;
    float* B0   = (float*)d_ws;
    float* B1   = B0 + NH;
    float* B2   = B1 + NH;
    float* numb = B2 + NH;
    float* denb = numb + NH;
    int*   mI   = (int*)(denb + NH);
    float* mask = (float*)(mI + NH);
    float* cnt  = mask + NN;
    float* outp = (float*)d_out;

    dim3 eb(HH, 2);
    const int nb_nh = (NN * HH) / 256;       // 25000 exactly
    const int nb_gemm = (NN + MT - 1) / MT;  // 782

    // pathway mask
    zero_f_kernel<<<(NN + 255) / 256, 256, 0, stream>>>(mask, NN);
    set_mask_kernel<<<(EPN + 255) / 256, 256, 0, stream>>>(pdst, mask);
    // node encoder -> B0
    enc_kernel<<<nb_nh, 256, 0, stream>>>(x, W_enc, b_enc, B0);

    // layer 0 gene conv: B0 -> B1
    init_aggr_kernel<<<nb_nh, 256, 0, stream>>>(mI, numb, denb);
    gene_edge_kernel<<<EE / 2, eb, 0, stream>>>(0, B0, edge_attr, W_edge, b_edge, src, dst, tg, mI, numb, denb);
    gene_edge_kernel<<<EE / 2, eb, 0, stream>>>(1, B0, edge_attr, W_edge, b_edge, src, dst, tg, mI, numb, denb);
    gemm_update_kernel<<<nb_gemm, 256, 0, stream>>>(B0, numb, denb, Wg, bg, nullptr, nullptr, B1);
    // h lives in B1 from here on (restored to B1 at end of each loop iteration)

    for (int l = 1; l < NLAY; l++) {
        // h2 = relu(LN(h)) -> B2
        ln_kernel<<<NN, HH, 0, stream>>>(B1, B2, gamma + (l - 1) * HH, beta + (l - 1) * HH, 1);
        // gene conv on B2 -> B0
        init_aggr_kernel<<<nb_nh, 256, 0, stream>>>(mI, numb, denb);
        gene_edge_kernel<<<EE / 2, eb, 0, stream>>>(0, B2, edge_attr, W_edge, b_edge, src, dst, tg + l, mI, numb, denb);
        gene_edge_kernel<<<EE / 2, eb, 0, stream>>>(1, B2, edge_attr, W_edge, b_edge, src, dst, tg + l, mI, numb, denb);
        gemm_update_kernel<<<nb_gemm, 256, 0, stream>>>(B2, numb, denb,
            Wg + (size_t)l * HH * HH, bg + l * HH, nullptr, nullptr, B0);
        // pathway conv on B0, masked residual -> B1
        init_aggr_kernel<<<nb_nh, 256, 0, stream>>>(mI, numb, denb);
        pw_edge_kernel<<<EPN / 2, eb, 0, stream>>>(0, B0, p_edge_attr, psrc, pdst, tp + l, mI, numb, denb);
        pw_edge_kernel<<<EPN / 2, eb, 0, stream>>>(1, B0, p_edge_attr, psrc, pdst, tp + l, mI, numb, denb);
        gemm_update_kernel<<<nb_gemm, 256, 0, stream>>>(B0, numb, denb,
            Wp + (size_t)l * HH * HH, bp + l * HH, mask, B0, B1);
    }

    // final LN -> B2
    ln_kernel<<<NN, HH, 0, stream>>>(B1, B2, gamma + (NLAY - 1) * HH, beta + (NLAY - 1) * HH, 0);

    // global mean pool into d_out
    zero_f_kernel<<<(GG * HH + 255) / 256, 256, 0, stream>>>(outp, GG * HH);
    zero_f_kernel<<<1, 256, 0, stream>>>(cnt, GG);
    pool_kernel<<<NN, HH, 0, stream>>>(B2, batch, outp, cnt);
    pool_div_kernel<<<(GG * HH + 255) / 256, 256, 0, stream>>>(outp, cnt);
}
